// SiGNN_89687507076358
// MI455X (gfx1250) — compile-verified, matrix operand order checked by reading, not run on hardware
//
#include <hip/hip_runtime.h>

typedef __bf16 bf16_t;
typedef __attribute__((ext_vector_type(4)))  __bf16 v4bf;
typedef __attribute__((ext_vector_type(8)))  __bf16 v8bf;
typedef __attribute__((ext_vector_type(16))) __bf16 v16bf;
typedef __attribute__((ext_vector_type(8)))  float  v8f;
typedef int gv4i __attribute__((vector_size(16)));   // matches builtin param type

// ---- problem sizes (from reference) ----
#define V_   100000
#define F_   512
#define N_   16384
#define H0_  256
#define H1_  128
#define C_   40
#define T_   6
#define S1_  5
#define S2_  2
#define M1_  (N_ * (1 + S1_))   // 98304 rows in layer-1
#define K1_  (2 * F_)           // 1024 (self|neigh concat)
#define N1_  (2 * H0_)          // 512  (t-half | x-half)
#define K2_  (2 * H0_)          // 512
#define N2_  (2 * H1_)          // 256

#if __has_builtin(__builtin_amdgcn_global_load_async_to_lds_b128)
#define USE_ASYNC_LDS 1
#else
#define USE_ASYNC_LDS 0
#endif

__device__ __forceinline__ float sigmoidf_(float x) {
    return 1.0f / (1.0f + __expf(-x));
}

#if USE_ASYNC_LDS
__device__ __forceinline__ void async_cp16(const bf16_t* g, bf16_t* l) {
    // global_load_async_to_lds_b128 (tracked by ASYNCcnt)
    __builtin_amdgcn_global_load_async_to_lds_b128((gv4i*)g, (gv4i*)l, 0, 0);
}
__device__ __forceinline__ void wait_async0() {
#if __has_builtin(__builtin_amdgcn_s_wait_asynccnt)
    __builtin_amdgcn_s_wait_asynccnt(0);
#else
    asm volatile("s_wait_asynccnt 0x0" ::: "memory");
#endif
}
#endif

// ---------------- init / pack kernels ----------------

__global__ __launch_bounds__(256)
void zero_f32_kernel(float* __restrict__ p, long long n) {
    long long i = (long long)blockIdx.x * 256 + threadIdx.x;
    if (i < n) p[i] = 0.0f;
}

__global__ __launch_bounds__(256)
void eacc_init_kernel(float* __restrict__ e, const float* __restrict__ b1,
                      const float* __restrict__ b2, const float* __restrict__ b3) {
    int i = blockIdx.x * 256 + threadIdx.x;
    if (i < N_ * H1_) {
        int h = i & (H1_ - 1);
        e[i] = b1[h] + b2[h] + b3[h];
    }
}

// agg1_w (3,4,H0,F) -> W1pack (3, 512, 1024) bf16, row r<256: out_t (parts 2|3), r>=256: out_x (parts 0|1)
__global__ __launch_bounds__(256)
void pack_w1_kernel(const float* __restrict__ w, bf16_t* __restrict__ W) {
    int i = blockIdx.x * 256 + threadIdx.x;
    if (i >= 3 * N1_ * K1_) return;
    int c = i / (N1_ * K1_);
    int r = (i / K1_) % N1_;
    int k = i % K1_;
    int h = r & (H0_ - 1);
    int f = k & (F_ - 1);
    int part = (r < H0_) ? ((k < F_) ? 2 : 3) : ((k < F_) ? 0 : 1);
    W[i] = (bf16_t)w[((size_t)(c * 4 + part) * H0_ + h) * F_ + f];
}

__global__ __launch_bounds__(256)
void pack_b1_kernel(const float* __restrict__ b, float* __restrict__ B) {
    int i = blockIdx.x * 256 + threadIdx.x;
    if (i >= 3 * N1_) return;
    int c = i / N1_, r = i % N1_, h = r & (H0_ - 1);
    int p0 = (r < H0_) ? 2 : 0, p1 = (r < H0_) ? 3 : 1;
    B[i] = b[(c * 4 + p0) * H0_ + h] + b[(c * 4 + p1) * H0_ + h];
}

// agg2_w (3,4,H1,H0) -> W2pack (3, 256, 512)
__global__ __launch_bounds__(256)
void pack_w2_kernel(const float* __restrict__ w, bf16_t* __restrict__ W) {
    int i = blockIdx.x * 256 + threadIdx.x;
    if (i >= 3 * N2_ * K2_) return;
    int c = i / (N2_ * K2_);
    int r = (i / K2_) % N2_;
    int k = i % K2_;
    int h = r & (H1_ - 1);
    int f = k & (H0_ - 1);
    int part = (r < H1_) ? ((k < H0_) ? 2 : 3) : ((k < H0_) ? 0 : 1);
    W[i] = (bf16_t)w[((size_t)(c * 4 + part) * H1_ + h) * H0_ + f];
}

__global__ __launch_bounds__(256)
void pack_b2_kernel(const float* __restrict__ b, float* __restrict__ B) {
    int i = blockIdx.x * 256 + threadIdx.x;
    if (i >= 3 * N2_) return;
    int c = i / N2_, r = i % N2_, h = r & (H1_ - 1);
    int p0 = (r < H1_) ? 2 : 0, p1 = (r < H1_) ? 3 : 1;
    B[i] = b[(c * 4 + p0) * H1_ + h] + b[(c * 4 + p1) * H1_ + h];
}

// ---------------- gather + neighbor-mean (channel independent, per t) ----------------
// A1 row i: [ self(512) | neigh_mean(512) ] in bf16
__global__ __launch_bounds__(128)
void build_a1_kernel(const float* __restrict__ x, const int* __restrict__ nodes,
                     const int* __restrict__ nbr1t, const int* __restrict__ nbr2t,
                     bf16_t* __restrict__ A1) {
    const int row = blockIdx.x;
    const int c4  = threadIdx.x * 4;
    float sx, sy, sz, sw;
    float ax = 0.f, ay = 0.f, az = 0.f, aw = 0.f, scale;
    if (row < N_) {
        const int si = nodes[row];
        const float4 sv = *(const float4*)(x + (size_t)si * F_ + c4);
        sx = sv.x; sy = sv.y; sz = sv.z; sw = sv.w;
#pragma unroll
        for (int j = 0; j < S1_; j++) {
            const int ni = nbr1t[row * S1_ + j];
            const float4 nv = *(const float4*)(x + (size_t)ni * F_ + c4);
            ax += nv.x; ay += nv.y; az += nv.z; aw += nv.w;
        }
        scale = 1.0f / (float)S1_;
    } else {
        const int r2 = row - N_;
        const int si = nbr1t[r2];
        const float4 sv = *(const float4*)(x + (size_t)si * F_ + c4);
        sx = sv.x; sy = sv.y; sz = sv.z; sw = sv.w;
#pragma unroll
        for (int j = 0; j < S2_; j++) {
            const int ni = nbr2t[r2 * S2_ + j];
            const float4 nv = *(const float4*)(x + (size_t)ni * F_ + c4);
            ax += nv.x; ay += nv.y; az += nv.z; aw += nv.w;
        }
        scale = 1.0f / (float)S2_;
    }
    v4bf so; so[0] = (bf16_t)sx; so[1] = (bf16_t)sy; so[2] = (bf16_t)sz; so[3] = (bf16_t)sw;
    *(v4bf*)(A1 + (size_t)row * K1_ + c4) = so;
    v4bf no; no[0] = (bf16_t)(ax * scale); no[1] = (bf16_t)(ay * scale);
    no[2] = (bf16_t)(az * scale); no[3] = (bf16_t)(aw * scale);
    *(v4bf*)(A1 + (size_t)row * K1_ + F_ + c4) = no;
}

// ---------------- WMMA bf16 GEMM:  C[M x Ncols] = A[M x K] * B^T  + bias ----------------
// B stored row-major (Ncols x K): B row n == output column n (B-frag loads mirror A-frag loads).
#define TM   128
#define TN   128
#define TK   32
#define LDSK 40   // 32 + 8 bf16 pad to stagger LDS banks (keeps 16B alignment)

__device__ __forceinline__ v16bf frag16(const bf16_t* p) {
    v8bf lo = *(const v8bf*)p;
    v8bf hi = *(const v8bf*)(p + 8);
    return __builtin_shufflevector(lo, hi, 0, 1, 2, 3, 4, 5, 6, 7,
                                   8, 9, 10, 11, 12, 13, 14, 15);
}

__global__ __launch_bounds__(256)
void gemm_bf16_wmma_kernel(const bf16_t* __restrict__ A,
                           const bf16_t* __restrict__ B,
                           const float*  __restrict__ bias,
                           float* __restrict__ Cmat,
                           int K, int ldc) {
    const int tid  = threadIdx.x;
    const int lane = tid & 31;          // wave32
    const int wave = tid >> 5;          // 8 waves
    const int wm   = wave & 3;          // 4 waves along M (32 rows each)
    const int wn   = wave >> 2;         // 2 waves along N (64 cols each)
    const size_t rowA0 = (size_t)blockIdx.y * TM;
    const size_t rowB0 = (size_t)blockIdx.x * TN;

    // global->LDS staging: each thread moves 4 x 16B per tile
    const int ldRow = tid >> 2;         // 0..63
    const int ldCol = (tid & 3) * 8;    // 0,8,16,24

    const bf16_t* gA = A + (rowA0 + ldRow) * (size_t)K + ldCol;
    const bf16_t* gB = B + (rowB0 + ldRow) * (size_t)K + ldCol;
    const size_t row64 = (size_t)64 * K;

    const int hk = (lane >> 4) * 16;    // K half selected by lane group
    const int rr = lane & 15;           // fragment row/col within tile

    v8f acc[2][4] = {};
    const int kt = K / TK;

#if USE_ASYNC_LDS
    // double-buffered LDS filled by async copies (ASYNCcnt), one barrier per K-step
    __shared__ bf16_t sA[2][TM * LDSK];
    __shared__ bf16_t sB[2][TN * LDSK];

    auto issue = [&](int kb, int buf) {
        const bf16_t* pa = gA + kb * TK;
        const bf16_t* pb = gB + kb * TK;
        async_cp16(pa,         &sA[buf][ldRow * LDSK + ldCol]);
        async_cp16(pa + row64, &sA[buf][(ldRow + 64) * LDSK + ldCol]);
        async_cp16(pb,         &sB[buf][ldRow * LDSK + ldCol]);
        async_cp16(pb + row64, &sB[buf][(ldRow + 64) * LDSK + ldCol]);
    };

    issue(0, 0);
    wait_async0();
    __syncthreads();

    int buf = 0;
    for (int kb = 0; kb < kt; ++kb) {
        if (kb + 1 < kt) issue(kb + 1, buf ^ 1);   // fill other buffer while we compute
        v16bf af[2], bfr[4];
#pragma unroll
        for (int i = 0; i < 2; i++)
            af[i] = frag16(&sA[buf][(wm * 32 + i * 16 + rr) * LDSK + hk]);
#pragma unroll
        for (int j = 0; j < 4; j++)
            bfr[j] = frag16(&sB[buf][(wn * 64 + j * 16 + rr) * LDSK + hk]);
#pragma unroll
        for (int i = 0; i < 2; i++)
#pragma unroll
            for (int j = 0; j < 4; j++)
                acc[i][j] = __builtin_amdgcn_wmma_f32_16x16x32_bf16(
                    false, af[i], false, bfr[j], (short)0, acc[i][j], false, false);
        if (kb + 1 < kt) {
            wait_async0();      // own async fills of next buffer done
            __syncthreads();    // everyone's fills done; everyone's reads of 'buf' done
        }
        buf ^= 1;
    }
#else
    // register-prefetch software pipeline fallback
    __shared__ bf16_t sA1[TM * LDSK];
    __shared__ bf16_t sB1[TN * LDSK];

    v8bf ra0, ra1, rb0, rb1;
    auto gload = [&](int kb) {
        const bf16_t* pa = gA + kb * TK;
        const bf16_t* pb = gB + kb * TK;
        ra0 = *(const v8bf*)pa;
        ra1 = *(const v8bf*)(pa + row64);
        rb0 = *(const v8bf*)pb;
        rb1 = *(const v8bf*)(pb + row64);
    };
    gload(0);
    for (int kb = 0; kb < kt; ++kb) {
        *(v8bf*)&sA1[ldRow * LDSK + ldCol]        = ra0;
        *(v8bf*)&sA1[(ldRow + 64) * LDSK + ldCol] = ra1;
        *(v8bf*)&sB1[ldRow * LDSK + ldCol]        = rb0;
        *(v8bf*)&sB1[(ldRow + 64) * LDSK + ldCol] = rb1;
        __syncthreads();
        if (kb + 1 < kt) gload(kb + 1);   // overlap next global loads with compute
        v16bf af[2], bfr[4];
#pragma unroll
        for (int i = 0; i < 2; i++)
            af[i] = frag16(&sA1[(wm * 32 + i * 16 + rr) * LDSK + hk]);
#pragma unroll
        for (int j = 0; j < 4; j++)
            bfr[j] = frag16(&sB1[(wn * 64 + j * 16 + rr) * LDSK + hk]);
#pragma unroll
        for (int i = 0; i < 2; i++)
#pragma unroll
            for (int j = 0; j < 4; j++)
                acc[i][j] = __builtin_amdgcn_wmma_f32_16x16x32_bf16(
                    false, af[i], false, bfr[j], (short)0, acc[i][j], false, false);
        if (kb + 1 < kt) __syncthreads();
    }
#endif

    // C/D layout: lanes 0-15 -> M=r, N=lane ; lanes 16-31 -> M=r+8, N=lane-16
    const int nloc = lane & 15;
    const int rb   = (lane >> 4) * 8;
#pragma unroll
    for (int i = 0; i < 2; i++) {
        const size_t row0 = rowA0 + wm * 32 + i * 16 + rb;
#pragma unroll
        for (int j = 0; j < 4; j++) {
            const int col  = (int)rowB0 + wn * 64 + j * 16 + nloc;
            const float bb = bias[col];
            float* cp = Cmat + row0 * (size_t)ldc + col;
#pragma unroll
            for (int r = 0; r < 8; r++) cp[(size_t)r * ldc] = acc[i][j][r] + bb;
        }
    }
}

// ---------------- LIF epilogues ----------------

__global__ __launch_bounds__(256)
void blif1_kernel(const float* __restrict__ C1, const float* __restrict__ snnw,
                  float* __restrict__ v1c, bf16_t* __restrict__ outs) {
    size_t i = (size_t)blockIdx.x * 256 + threadIdx.x;
    if (i >= (size_t)M1_ * H0_) return;
    int h = (int)(i & (H0_ - 1));
    size_t row = i >> 8;
    float outt = C1[row * (size_t)N1_ + h];
    float outx = sigmoidf_(C1[row * (size_t)N1_ + H0_ + h]);
    float sw = sigmoidf_(snnw[h]);
    float v = sw * v1c[i] + outt;
    float s = (v >= 1.0f) ? 1.0f : 0.0f;   // VTH = 1.0
    v1c[i] = v - s;
    outs[i] = (bf16_t)(outx * s);
}

// A2 row n: [ outs[n] (256) | mean_j outs[N + n*5 + j] (256) ]
__global__ __launch_bounds__(256)
void build_a2_kernel(const bf16_t* __restrict__ outs, bf16_t* __restrict__ A2) {
    int i = blockIdx.x * 256 + threadIdx.x;
    if (i >= N_ * H0_) return;
    int q = i & (H0_ - 1);
    int n = i >> 8;
    A2[(size_t)n * K2_ + q] = outs[(size_t)n * H0_ + q];
    float acc = 0.f;
#pragma unroll
    for (int j = 0; j < S1_; j++)
        acc += (float)outs[(size_t)(N_ + n * S1_ + j) * H0_ + q];
    A2[(size_t)n * K2_ + H0_ + q] = (bf16_t)(acc * (1.0f / (float)S1_));
}

__global__ __launch_bounds__(256)
void blif2_pool_kernel(const float* __restrict__ C2, const float* __restrict__ snnw,
                       float* __restrict__ v2c, float* __restrict__ eacc,
                       const float* __restrict__ pw, int pwStride, int pwIdx) {
    int i = blockIdx.x * 256 + threadIdx.x;
    if (i >= N_ * H1_) return;
    int h = i & (H1_ - 1);
    int n = i >> 7;
    float outt = C2[(size_t)n * N2_ + h];
    float outx = sigmoidf_(C2[(size_t)n * N2_ + H1_ + h]);
    float sw = sigmoidf_(snnw[h]);
    float v = sw * v2c[i] + outt;
    float s = (v >= 1.0f) ? 1.0f : 0.0f;
    v2c[i] = v - s;
    eacc[i] += (outx * s) * pw[h * pwStride + pwIdx];
}

__global__ __launch_bounds__(256)
void final_kernel(const float* __restrict__ eacc, const float* __restrict__ mw,
                  const float* __restrict__ mb, float* __restrict__ out) {
    int i = blockIdx.x * 256 + threadIdx.x;
    if (i >= N_ * C_) return;
    int cc = i % C_;
    int n  = i / C_;
    const float* e = eacc + (size_t)n * H1_;
    const float* w = mw + (size_t)cc * H1_;
    float s = 0.f;
#pragma unroll 8
    for (int k = 0; k < H1_; k++) s += e[k] * w[k];
    out[i] = s * (1.0f / 3.0f) + mb[cc];
}

// ---------------- host orchestration ----------------

extern "C" void kernel_launch(void* const* d_in, const int* in_sizes, int n_in,
                              void* d_out, int out_size, void* d_ws, size_t ws_size,
                              hipStream_t stream) {
    const float* x       = (const float*)d_in[0];
    const float* agg1_w  = (const float*)d_in[1];
    const float* agg1_b  = (const float*)d_in[2];
    const float* agg2_w  = (const float*)d_in[3];
    const float* agg2_b  = (const float*)d_in[4];
    const float* snn1_w  = (const float*)d_in[5];
    const float* snn2_w  = (const float*)d_in[6];
    const float* pool_w1 = (const float*)d_in[7];
    const float* pool_b1 = (const float*)d_in[8];
    const float* pool_w2 = (const float*)d_in[9];
    const float* pool_b2 = (const float*)d_in[10];
    const float* pool_w3 = (const float*)d_in[11];
    const float* pool_b3 = (const float*)d_in[12];
    const float* mtg_w   = (const float*)d_in[13];
    const float* mtg_b   = (const float*)d_in[14];
    const int*   nodes   = (const int*)d_in[15];
    const int*   nbr1    = (const int*)d_in[16];
    const int*   nbr2    = (const int*)d_in[17];
    float* out = (float*)d_out;

    // workspace carve-up (256B aligned)
    char* base = (char*)d_ws;
    size_t off = 0;
    auto take = [&](size_t bytes) -> char* {
        char* p = base + off;
        off = (off + bytes + 255) & ~(size_t)255;
        return p;
    };
    bf16_t* W1p  = (bf16_t*)take((size_t)3 * N1_ * K1_ * sizeof(bf16_t));
    float*  B1p  = (float*) take((size_t)3 * N1_ * sizeof(float));
    bf16_t* W2p  = (bf16_t*)take((size_t)3 * N2_ * K2_ * sizeof(bf16_t));
    float*  B2p  = (float*) take((size_t)3 * N2_ * sizeof(float));
    float*  v1   = (float*) take((size_t)3 * M1_ * H0_ * sizeof(float));
    float*  v2   = (float*) take((size_t)3 * N_ * H1_ * sizeof(float));
    float*  eacc = (float*) take((size_t)N_ * H1_ * sizeof(float));
    bf16_t* A1   = (bf16_t*)take((size_t)M1_ * K1_ * sizeof(bf16_t));
    float*  C1   = (float*) take((size_t)M1_ * N1_ * sizeof(float));
    bf16_t* outs = (bf16_t*)take((size_t)M1_ * H0_ * sizeof(bf16_t));
    bf16_t* A2   = (bf16_t*)take((size_t)N_ * K2_ * sizeof(bf16_t));
    float*  C2   = (float*) take((size_t)N_ * N2_ * sizeof(float));

    // pack weights / biases
    pack_w1_kernel<<<(3 * N1_ * K1_ + 255) / 256, 256, 0, stream>>>(agg1_w, W1p);
    pack_b1_kernel<<<(3 * N1_ + 255) / 256, 256, 0, stream>>>(agg1_b, B1p);
    pack_w2_kernel<<<(3 * N2_ * K2_ + 255) / 256, 256, 0, stream>>>(agg2_w, W2p);
    pack_b2_kernel<<<(3 * N2_ + 255) / 256, 256, 0, stream>>>(agg2_b, B2p);

    // zero LIF state, init pooled accumulator with summed biases
    {
        long long nv1 = (long long)3 * M1_ * H0_;
        long long nv2 = (long long)3 * N_ * H1_;
        zero_f32_kernel<<<(unsigned)((nv1 + 255) / 256), 256, 0, stream>>>(v1, nv1);
        zero_f32_kernel<<<(unsigned)((nv2 + 255) / 256), 256, 0, stream>>>(v2, nv2);
        eacc_init_kernel<<<(N_ * H1_ + 255) / 256, 256, 0, stream>>>(eacc, pool_b1, pool_b2, pool_b3);
    }

    for (int t = 0; t < T_; ++t) {
        build_a1_kernel<<<M1_, 128, 0, stream>>>(
            x, nodes, nbr1 + (size_t)t * N_ * S1_, nbr2 + (size_t)t * N_ * S1_ * S2_, A1);

        struct Step { int c; const float* pw; int stride; int idx; } steps[3];
        int ns = 0;
        steps[ns++] = {0, pool_w1, T_, t};
        if (t % 2 == 0)       steps[ns++] = {1, pool_w2, T_ / 2 + T_ % 2, t / 2};
        if ((t + 1) % 3 == 0) steps[ns++] = {2, pool_w3, T_ / 3 + T_ % 3, (t + 1) / 3 - 1};

        for (int si = 0; si < ns; ++si) {
            const int c = steps[si].c;
            // layer 1: (98304 x 1024) x (1024 x 512)
            gemm_bf16_wmma_kernel<<<dim3(N1_ / TN, M1_ / TM), 256, 0, stream>>>(
                A1, W1p + (size_t)c * N1_ * K1_, B1p + (size_t)c * N1_, C1, K1_, N1_);
            blif1_kernel<<<(M1_ * H0_) / 256, 256, 0, stream>>>(
                C1, snn1_w + (size_t)c * H0_, v1 + (size_t)c * M1_ * H0_, outs);
            build_a2_kernel<<<(N_ * H0_) / 256, 256, 0, stream>>>(outs, A2);
            // layer 2: (16384 x 512) x (512 x 256)
            gemm_bf16_wmma_kernel<<<dim3(N2_ / TN, N_ / TM), 256, 0, stream>>>(
                A2, W2p + (size_t)c * N2_ * K2_, B2p + (size_t)c * N2_, C2, K2_, N2_);
            blif2_pool_kernel<<<(N_ * H1_) / 256, 256, 0, stream>>>(
                C2, snn2_w + (size_t)c * H1_, v2 + (size_t)c * N_ * H1_, eacc,
                steps[si].pw, steps[si].stride, steps[si].idx);
        }
    }

    final_kernel<<<(N_ * C_) / 256, 256, 0, stream>>>(eacc, mtg_w, mtg_b, out);
}
